// HolterFM_21157008900565
// MI455X (gfx1250) — matrix operand order repaired and checked by reading, not compiled, not run on hardware
//
#include <hip/hip_runtime.h>
#include <math.h>
#include <stdint.h>

// ---------------- problem constants ----------------
constexpr int B    = 8;
constexpr int M    = 8192;
constexpr int E    = 128;
constexpr int EPL  = 64;
constexpr long long NBM = (long long)B * M;      // 65536 rows
constexpr int BEAT_D = 256, EP_D = 384, RHY_D = 128, DAY_D = 512;
constexpr int TOK_H = 512, CODE_D = 64, N_CODES = 512;
constexpr int WIN_FLAT = 246;                    // 80*3 + 6

typedef __attribute__((ext_vector_type(16))) _Float16 v16h;
typedef __attribute__((ext_vector_type(8)))  _Float16 v8h;
typedef __attribute__((ext_vector_type(8)))  float    v8f;
typedef __attribute__((ext_vector_type(4)))  unsigned int v4u;
typedef __attribute__((ext_vector_type(8)))  int      v8i;
typedef __attribute__((ext_vector_type(4)))  int      v4i;

__device__ __forceinline__ float gelu_tanh(float x) {
    const float c = 0.7978845608028654f;           // sqrt(2/pi)
    float t = tanhf(c * (x + 0.044715f * x * x * x));
    return 0.5f * x * (1.0f + t);
}

#if __has_builtin(__builtin_amdgcn_tensor_load_to_lds) && \
    __has_builtin(__builtin_amdgcn_s_wait_tensorcnt)
#define HAVE_TDM 1
#else
#define HAVE_TDM 0
#endif

#if HAVE_TDM
// Issue a TDM DMA of a 16-row x 32-col f16 tile (row stride = lda elements)
// from global memory into LDS at byte offset lds_off.  D# per CDNA5 ISA ch.8:
// group0: count=1, lds_addr, global_addr, type=2("image")
// group1: data_size=2B, tensor_dim0=lda, tensor_dim1=1M, tile=32x16, stride0=lda
__device__ __forceinline__ void tdm_load_tile16x32(unsigned lds_off,
                                                   const _Float16* gptr, int lda)
{
    unsigned long long ga = (unsigned long long)(uintptr_t)gptr;
    v4u g0;
    g0[0] = 1u;                                            // count=1, user mode
    g0[1] = lds_off;                                       // lds_addr
    g0[2] = (unsigned)ga;                                  // global_addr[31:0]
    g0[3] = (unsigned)((ga >> 32) & 0x01FFFFFFu) | (2u << 30); // addr[56:32]|type=2
    v8i g1;
    g1[0] = 1 << 16;                                       // data_size = 2 bytes
    g1[1] = (lda & 0xFFFF) << 16;                          // tensor_dim0[15:0]
    g1[2] = (lda >> 16) & 0xFFFF;                          // tensor_dim0[31:16], dim1_lo=0
    g1[3] = 16 | (32 << 16);                               // tensor_dim1[31:16]=16 (1M rows), tile_dim0=32
    g1[4] = 16;                                            // tile_dim1=16, tile_dim2=0
    g1[5] = lda;                                           // tensor_dim0_stride[31:0]
    g1[6] = 0;
    g1[7] = 0;
    v4i z4 = {0, 0, 0, 0};
    v8i z8 = {0, 0, 0, 0, 0, 0, 0, 0};
    // 6-arg form (clang-23 / therock headers): g0, g1, g2, g3, g4, cpol
    __builtin_amdgcn_tensor_load_to_lds(g0, g1, z4, z4, z8, 0);
}
#endif

// =====================================================================
// WMMA GEMM:  C[rows,N] = act(A[rows,Kpad] * W + bias)
//   A : f16 row-major, leading dim lda (>= Kpad, both mult of 32)
//   WT: f16, W transposed+padded: WT[n*Kpad + k]
//   C : f32 or f16, leading dim ldc, column offset coloff
// Block = 128 threads = 4 waves; wave -> one 16x16 tile; block -> 64 cols.
// A tile staged into LDS by the Tensor Data Mover (fallback: coop copy).
// =====================================================================
template <bool GELU, bool OUT_F16>
__global__ __launch_bounds__(128) void gemm_wmma(
    const _Float16* __restrict__ A, int lda, int Kpad,
    const _Float16* __restrict__ WT,
    const float* __restrict__ bias,
    void* __restrict__ Cout, int ldc, int coloff)
{
    __shared__ __align__(16) _Float16 As[16 * 32];

    const int wave    = threadIdx.x >> 5;
    const int lane    = threadIdx.x & 31;
    const int row0    = blockIdx.x * 16;
    const int m       = lane & 15;
    const int halfsel = lane >> 4;
    const int n_local = blockIdx.y * 64 + wave * 16 + (lane & 15);

#if HAVE_TDM
    const unsigned lds_off = (unsigned)(unsigned long long)(uintptr_t)&As[0];
#endif

    v8f acc;
#pragma unroll
    for (int i = 0; i < 8; ++i) acc[i] = 0.0f;

    const _Float16* wrow = WT + (long long)n_local * Kpad;

    for (int k0 = 0; k0 < Kpad; k0 += 32) {
        __syncthreads();                       // LDS reuse from prev iteration
#if HAVE_TDM
        if (threadIdx.x < 32) {                // wave 0 issues the DMA
            tdm_load_tile16x32(lds_off, A + (long long)row0 * lda + k0, lda);
            __builtin_amdgcn_s_wait_tensorcnt(0);
        }
#else
        {
            int e  = threadIdx.x * 4;
            int r  = e >> 5;
            int kk = e & 31;
            const uint2* src = reinterpret_cast<const uint2*>(
                A + (long long)(row0 + r) * lda + k0 + kk);
            *reinterpret_cast<uint2*>(&As[r * 32 + kk]) = *src;
        }
#endif
        __syncthreads();

        // A fragment: two contiguous 16B chunks of row m (ISA 16-bit A layout)
        union { v16h v; v8h h[2]; } au;
        au.h[0] = *reinterpret_cast<const v8h*>(&As[m * 32 + halfsel * 8]);
        au.h[1] = *reinterpret_cast<const v8h*>(&As[m * 32 + 16 + halfsel * 8]);

        // B fragment: two 16B chunks of column n (pre-transposed W), same k order as A
        union { v16h v; v8h h[2]; } bu;
        bu.h[0] = *reinterpret_cast<const v8h*>(wrow + k0 + halfsel * 8);
        bu.h[1] = *reinterpret_cast<const v8h*>(wrow + k0 + 16 + halfsel * 8);
        v16h b = bu.v;

        __builtin_prefetch(wrow + k0 + 32, 0, 1);  // global_prefetch_b8

        acc = __builtin_amdgcn_wmma_f32_16x16x32_f16(
            false, au.v, false, b, (short)0, acc, false, false);
    }

    float bn = bias ? bias[n_local] : 0.0f;
#pragma unroll
    for (int r = 0; r < 8; ++r) {
        int mm = r + 8 * halfsel;
        float v = acc[r] + bn;
        if (GELU) v = gelu_tanh(v);
        long long off = (long long)(row0 + mm) * ldc + coloff + n_local;
        if (OUT_F16) ((_Float16*)Cout)[off] = (_Float16)v;
        else         ((float*)Cout)[off]    = v;
    }
}

// ---------------- weight pack: f32 (Kreal x N) -> f16 transposed (N x Kpad) ----------------
__global__ void pack_w(const float* __restrict__ W, _Float16* __restrict__ WT,
                       int Kreal, int Kpad, int N)
{
    long long idx = (long long)blockIdx.x * blockDim.x + threadIdx.x;
    if (idx >= (long long)N * Kpad) return;
    int n = (int)(idx / Kpad);
    int k = (int)(idx % Kpad);
    float v = (k < Kreal) ? W[(long long)k * N + n] : 0.0f;
    WT[idx] = (_Float16)v;
}

// ---------------- pack tokenizer input, zero-padded 246 -> 256 ----------------
__global__ void pack_x(const float* __restrict__ win, const float* __restrict__ meta,
                       _Float16* __restrict__ xpad)
{
    long long idx = (long long)blockIdx.x * blockDim.x + threadIdx.x;
    if (idx >= NBM * 256) return;
    int row = (int)(idx >> 8);
    int c   = (int)(idx & 255);
    float v = 0.0f;
    if (c < 240)           v = win[(long long)row * 240 + c];
    else if (c < WIN_FLAT) v = meta[(long long)row * 6 + (c - 240)];
    xpad[idx] = (_Float16)v;
}

// ---------------- VQ: argmin over 512 codes + vq_loss ----------------
__global__ void zero_scalar(float* p) { if (threadIdx.x == 0 && blockIdx.x == 0) *p = 0.0f; }

__global__ void vq_kernel(const float* __restrict__ z, const float* __restrict__ cb,
                          int* __restrict__ ci, float* __restrict__ ci_f,
                          float* __restrict__ vql)
{
    int row = blockIdx.x * blockDim.x + threadIdx.x;
    if (row >= (int)NBM) return;
    float zr[CODE_D];
#pragma unroll
    for (int i = 0; i < CODE_D; ++i) zr[i] = z[(long long)row * CODE_D + i];
    float best = 3.4e38f;
    int   bi   = 0;
    for (int c = 0; c < N_CODES; ++c) {
        float d = 0.0f;
#pragma unroll 8
        for (int i = 0; i < CODE_D; ++i) {
            float t = zr[i] - cb[c * CODE_D + i];
            d += t * t;
        }
        if (d < best) { best = d; bi = c; }
    }
    ci[row]   = bi;
    ci_f[row] = (float)bi;
    atomicAdd(vql, best * (1.0f / ((float)NBM * (float)CODE_D)));
}

// ---------------- rhythm branch ----------------
__global__ void rhythm_kernel(const int* __restrict__ ci, const int* __restrict__ rr,
                              const int* __restrict__ rrp,
                              const float* __restrict__ hs, const float* __restrict__ hc,
                              const float* __restrict__ ce, const float* __restrict__ re,
                              const float* __restrict__ rpe, const float* __restrict__ hw,
                              _Float16* __restrict__ brhy)
{
    long long idx = (long long)blockIdx.x * blockDim.x + threadIdx.x;
    if (idx >= NBM * RHY_D) return;
    int row = (int)(idx >> 7);
    int d   = (int)(idx & 127);
    float v = ce[ci[row] * RHY_D + d] + re[rr[row] * RHY_D + d] +
              rpe[rrp[row] * RHY_D + d] +
              hs[row] * hw[d] + hc[row] * hw[RHY_D + d];
    brhy[idx] = (_Float16)gelu_tanh(v);
}

// ---------------- per-episode means (ctx 384-d, rhythm 128-d) ----------------
__global__ __launch_bounds__(128) void ep_reduce(
    const _Float16* __restrict__ bctx, const _Float16* __restrict__ brhy,
    _Float16* __restrict__ epm, _Float16* __restrict__ fusedA)
{
    int be = blockIdx.x;                 // 0..1023
    int b  = be >> 7, e = be & 127;
    long long base = (long long)b * M + (long long)e * EPL;
    for (int d = threadIdx.x; d < EP_D; d += 128) {
        float s = 0.0f;
        for (int l = 0; l < EPL; ++l) s += (float)bctx[(base + l) * EP_D + d];
        epm[(long long)be * EP_D + d] = (_Float16)(s * (1.0f / EPL));
    }
    for (int d = threadIdx.x; d < RHY_D; d += 128) {
        float s = 0.0f;
        for (int l = 0; l < EPL; ++l) s += (float)brhy[(base + l) * RHY_D + d];
        fusedA[(long long)be * (EP_D + RHY_D) + EP_D + d] = (_Float16)(s * (1.0f / EPL));
    }
}

// ---------------- zero invalid episode rows of fusedA ----------------
__global__ void mask_fused(const int* __restrict__ nep, _Float16* __restrict__ fusedA)
{
    int idx = blockIdx.x * blockDim.x + threadIdx.x;   // 1024*512
    if (idx >= 1024 * 512) return;
    int row = idx >> 9;
    int b = row >> 7, e = row & 127;
    if (e >= nep[b]) fusedA[idx] = (_Float16)0.0f;
}

// ---------------- masked day mean over episodes ----------------
__global__ void day_mean(const float* __restrict__ epctx, const int* __restrict__ nep,
                         float* __restrict__ day)
{
    int idx = blockIdx.x * blockDim.x + threadIdx.x;   // 8*512
    if (idx >= B * DAY_D) return;
    int b = idx >> 9, d = idx & 511;
    int ne = nep[b];
    float s = 0.0f;
    for (int e = 0; e < ne; ++e) s += epctx[((long long)b * E + e) * DAY_D + d];
    float den = (ne > 0) ? (float)ne : 1.0f;
    day[idx] = s / den;
}

// ---------------- build [beat_embed | masked beat_ctx | beat_rhythm] ----------------
__global__ void pack_cat(const _Float16* __restrict__ be_, const _Float16* __restrict__ bctx,
                         const _Float16* __restrict__ brhy, const int* __restrict__ nep,
                         _Float16* __restrict__ cat)
{
    long long idx = (long long)blockIdx.x * blockDim.x + threadIdx.x;
    if (idx >= NBM * 768) return;
    int row = (int)(idx / 768);
    int c   = (int)(idx % 768);
    int b = row >> 13, mm = row & 8191;
    _Float16 v;
    if (c < 256)      v = be_[(long long)row * 256 + c];
    else if (c < 640) v = (mm < nep[b] * EPL) ? bctx[(long long)row * EP_D + (c - 256)]
                                              : (_Float16)0.0f;
    else              v = brhy[(long long)row * RHY_D + (c - 640)];
    cat[idx] = v;
}

// =====================================================================
extern "C" void kernel_launch(void* const* d_in, const int* in_sizes, int n_in,
                              void* d_out, int out_size, void* d_ws, size_t ws_size,
                              hipStream_t stream)
{
    (void)in_sizes; (void)n_in; (void)out_size; (void)ws_size;
    const float* windows  = (const float*)d_in[0];
    const float* metadata = (const float*)d_in[1];
    const int*   n_eps    = (const int*)d_in[3];
    const int*   rr_bins  = (const int*)d_in[4];
    const int*   rrp_bins = (const int*)d_in[5];
    const float* hour_sin = (const float*)d_in[6];
    const float* hour_cos = (const float*)d_in[7];
    const float* w_tok1 = (const float*)d_in[8];  const float* b_tok1 = (const float*)d_in[9];
    const float* w_tok2 = (const float*)d_in[10]; const float* b_tok2 = (const float*)d_in[11];
    const float* w_code = (const float*)d_in[12]; const float* codebook = (const float*)d_in[13];
    const float* w_ep1  = (const float*)d_in[14]; const float* b_ep1  = (const float*)d_in[15];
    const float* w_ep2  = (const float*)d_in[16]; const float* b_ep2  = (const float*)d_in[17];
    const float* w_eptk = (const float*)d_in[18]; const float* b_eptk = (const float*)d_in[19];
    const float* code_emb = (const float*)d_in[20];
    const float* rr_emb   = (const float*)d_in[21];
    const float* rrp_emb  = (const float*)d_in[22];
    const float* hour_w   = (const float*)d_in[23];
    const float* w_fuse = (const float*)d_in[24]; const float* b_fuse = (const float*)d_in[25];
    const float* w_day  = (const float*)d_in[26]; const float* b_day  = (const float*)d_in[27];
    const float* w_bp1  = (const float*)d_in[28]; const float* b_bp1  = (const float*)d_in[29];
    const float* w_bp2  = (const float*)d_in[30]; const float* b_bp2  = (const float*)d_in[31];

    // ---- workspace layout (256B aligned) ----
    char* ws = (char*)d_ws;
    size_t off = 0;
    auto alloc = [&](size_t bytes) { size_t p = off; off += (bytes + 255) & ~(size_t)255; return p; };
    size_t o_xpad = alloc(NBM * 256 * 2);   // cat overlays xpad+h1 later
    size_t o_h1   = alloc(NBM * 512 * 2);
    size_t o_beatE= alloc(NBM * 256 * 2);
    size_t o_z    = alloc(NBM * 64  * 4);
    size_t o_ci   = alloc(NBM * 4);
    size_t o_he   = alloc(NBM * 384 * 2);   // bph overlays he+bctx later
    size_t o_bctx = alloc(NBM * 384 * 2);
    size_t o_brhy = alloc(NBM * 128 * 2);
    size_t o_epm  = alloc(1024 * 384 * 2);
    size_t o_fA   = alloc(1024 * 512 * 2);
    size_t o_gf   = alloc(1024 * 512 * 2);
    // packed f16 transposed weights (N x Kpad each)
    size_t o_wt1  = alloc((size_t)512 * 256 * 2);
    size_t o_wt2  = alloc((size_t)256 * 512 * 2);
    size_t o_wc   = alloc((size_t)64  * 256 * 2);
    size_t o_we1  = alloc((size_t)384 * 256 * 2);
    size_t o_we2  = alloc((size_t)384 * 384 * 2);
    size_t o_wet  = alloc((size_t)384 * 384 * 2);
    size_t o_wf   = alloc((size_t)512 * 512 * 2);
    size_t o_wd   = alloc((size_t)512 * 512 * 2);
    size_t o_wb1  = alloc((size_t)512 * 768 * 2);
    size_t o_wb2  = alloc((size_t)512 * 512 * 2);

    _Float16* xpad  = (_Float16*)(ws + o_xpad);
    _Float16* h1    = (_Float16*)(ws + o_h1);
    _Float16* beatE = (_Float16*)(ws + o_beatE);
    float*    z     = (float*)   (ws + o_z);
    int*      ci    = (int*)     (ws + o_ci);
    _Float16* he    = (_Float16*)(ws + o_he);
    _Float16* bctx  = (_Float16*)(ws + o_bctx);
    _Float16* brhy  = (_Float16*)(ws + o_brhy);
    _Float16* epm   = (_Float16*)(ws + o_epm);
    _Float16* fusedA= (_Float16*)(ws + o_fA);
    _Float16* gf    = (_Float16*)(ws + o_gf);
    _Float16* cat   = (_Float16*)(ws + o_xpad);  // reuse: 100.7 MB span
    _Float16* bph   = (_Float16*)(ws + o_he);    // reuse: he+bctx span
    _Float16* wt1   = (_Float16*)(ws + o_wt1);
    _Float16* wt2   = (_Float16*)(ws + o_wt2);
    _Float16* wc    = (_Float16*)(ws + o_wc);
    _Float16* we1   = (_Float16*)(ws + o_we1);
    _Float16* we2   = (_Float16*)(ws + o_we2);
    _Float16* wet   = (_Float16*)(ws + o_wet);
    _Float16* wf    = (_Float16*)(ws + o_wf);
    _Float16* wd    = (_Float16*)(ws + o_wd);
    _Float16* wb1   = (_Float16*)(ws + o_wb1);
    _Float16* wb2   = (_Float16*)(ws + o_wb2);

    // ---- output layout (tuple concatenated, f32 view) ----
    float* out         = (float*)d_out;
    float* beat_repr   = out;                                    // (B,M,512)
    float* day_embed   = beat_repr + NBM * 512;                  // (B,512)
    float* episode_ctx = day_embed + B * DAY_D;                  // (B,E,512)
    float* code_idx_f  = episode_ctx + (long long)B * E * DAY_D; // (B,M)
    float* vq_loss     = code_idx_f + NBM;                       // scalar

    const int BT = 256;
    const int rows = (int)NBM;
    dim3 blk(128);
    auto wgrid = [&](long long total) { return (unsigned)((total + BT - 1) / BT); };

    // 0) pack inputs + all weights (f16, transposed, K padded)
    pack_x<<<wgrid(NBM * 256), BT, 0, stream>>>(windows, metadata, xpad);
    pack_w<<<wgrid(512 * 256), BT, 0, stream>>>(w_tok1, wt1, WIN_FLAT, 256, 512);
    pack_w<<<wgrid(256 * 512), BT, 0, stream>>>(w_tok2, wt2, 512, 512, 256);
    pack_w<<<wgrid(64  * 256), BT, 0, stream>>>(w_code, wc,  256, 256, 64);
    pack_w<<<wgrid(384 * 256), BT, 0, stream>>>(w_ep1,  we1, 256, 256, 384);
    pack_w<<<wgrid(384 * 384), BT, 0, stream>>>(w_ep2,  we2, 384, 384, 384);
    pack_w<<<wgrid(384 * 384), BT, 0, stream>>>(w_eptk, wet, 384, 384, 384);
    pack_w<<<wgrid(512 * 512), BT, 0, stream>>>(w_fuse, wf,  512, 512, 512);
    pack_w<<<wgrid(512 * 512), BT, 0, stream>>>(w_day,  wd,  512, 512, 512);
    pack_w<<<wgrid(512 * 768), BT, 0, stream>>>(w_bp1,  wb1, 768, 768, 512);
    pack_w<<<wgrid(512 * 512), BT, 0, stream>>>(w_bp2,  wb2, 512, 512, 512);

    // 1) tokenizer MLP: 246->512 (gelu) -> 256
    gemm_wmma<true,  true ><<<dim3(rows / 16, TOK_H / 64), blk, 0, stream>>>(
        xpad, 256, 256, wt1, b_tok1, h1, TOK_H, 0);
    gemm_wmma<false, true ><<<dim3(rows / 16, BEAT_D / 64), blk, 0, stream>>>(
        h1, 512, 512, wt2, b_tok2, beatE, BEAT_D, 0);

    // 2) z = beatE @ w_code ; VQ argmin + loss
    gemm_wmma<false, false><<<dim3(rows / 16, CODE_D / 64), blk, 0, stream>>>(
        beatE, 256, 256, wc, nullptr, z, CODE_D, 0);
    zero_scalar<<<1, 32, 0, stream>>>(vq_loss);
    vq_kernel<<<rows / BT, BT, 0, stream>>>(z, codebook, ci, code_idx_f, vq_loss);

    // 3) rhythm branch
    rhythm_kernel<<<wgrid(NBM * RHY_D), BT, 0, stream>>>(
        ci, rr_bins, rrp_bins, hour_sin, hour_cos,
        code_emb, rr_emb, rrp_emb, hour_w, brhy);

    // 4) episode encoder: 256->384 (gelu) -> 384
    gemm_wmma<true,  true ><<<dim3(rows / 16, EP_D / 64), blk, 0, stream>>>(
        beatE, 256, 256, we1, b_ep1, he, EP_D, 0);
    gemm_wmma<false, true ><<<dim3(rows / 16, EP_D / 64), blk, 0, stream>>>(
        he, 384, 384, we2, b_ep2, bctx, EP_D, 0);

    // 5) episode means, episode token GEMM into fusedA, masking
    ep_reduce<<<B * E, 128, 0, stream>>>(bctx, brhy, epm, fusedA);
    gemm_wmma<false, true ><<<dim3(B * E / 16, EP_D / 64), blk, 0, stream>>>(
        epm, 384, 384, wet, b_eptk, fusedA, EP_D + RHY_D, 0);
    mask_fused<<<(B * E * 512 + BT - 1) / BT, BT, 0, stream>>>(n_eps, fusedA);

    // 6) day encoder: fuse (gelu) -> day -> episode_ctx; masked day mean
    gemm_wmma<true,  true ><<<dim3(B * E / 16, DAY_D / 64), blk, 0, stream>>>(
        fusedA, 512, 512, wf, b_fuse, gf, DAY_D, 0);
    gemm_wmma<false, false><<<dim3(B * E / 16, DAY_D / 64), blk, 0, stream>>>(
        gf, 512, 512, wd, b_day, episode_ctx, DAY_D, 0);
    day_mean<<<(B * DAY_D + BT - 1) / BT, BT, 0, stream>>>(episode_ctx, n_eps, day_embed);

    // 7) beat projection: cat(768) -> 512 (gelu) -> 512
    pack_cat<<<wgrid(NBM * 768), BT, 0, stream>>>(beatE, bctx, brhy, n_eps, cat);
    gemm_wmma<true,  true ><<<dim3(rows / 16, DAY_D / 64), blk, 0, stream>>>(
        cat, 768, 768, wb1, b_bp1, bph, DAY_D, 0);
    gemm_wmma<false, false><<<dim3(rows / 16, DAY_D / 64), blk, 0, stream>>>(
        bph, 512, 512, wb2, b_bp2, beat_repr, DAY_D, 0);
}
